// MultiheadSelfAttention_71923522339086
// MI455X (gfx1250) — matrix operand order repaired and checked by reading, not compiled, not run on hardware
//
#include <hip/hip_runtime.h>
#include <hip/hip_bf16.h>

#define S_LEN   2048
#define DMODEL  1024
#define NHEADS  16
#define HDIM    64
#define BATCH   4
#define BS_ROWS (BATCH * S_LEN)   // 8192

typedef __bf16 bf16_t;
typedef __bf16 bf16x8  __attribute__((ext_vector_type(8)));
typedef __bf16 bf16x16 __attribute__((ext_vector_type(16)));
typedef float  f32x8   __attribute__((ext_vector_type(8)));

union FragU { bf16x16 v; bf16x8 h[2]; };

// A-matrix 16x32 bf16 fragment: lane = M row, element e -> K = hlf*16+e
// (contiguous, matches load_b_frag)
__device__ __forceinline__ bf16x16 load_a_frag(const bf16_t* base, int ld, int row0, int col0) {
  const int lane = threadIdx.x & 31;
  const int m    = lane & 15;
  const int hlf  = lane >> 4;
  const bf16_t* p = base + (size_t)(row0 + m) * ld + col0 + hlf * 16;
  FragU u;
  u.h[0] = *(const bf16x8*)(p);
  u.h[1] = *(const bf16x8*)(p + 8);
  return u.v;
}

// B-matrix 32x16 bf16 fragment: lane = N column (row of the stored matrix),
// elements = 16 contiguous K starting at hlf*16   (ISA 7.12.4 analogue)
__device__ __forceinline__ bf16x16 load_b_frag(const bf16_t* base, int ld, int row0, int col0) {
  const int lane = threadIdx.x & 31;
  const int n    = lane & 15;
  const int hlf  = lane >> 4;
  const bf16_t* p = base + (size_t)(row0 + n) * ld + col0 + hlf * 16;
  return *(const bf16x16*)(p);
}

__device__ __forceinline__ f32x8 wmma_bf16(bf16x16 a, bf16x16 b, f32x8 c) {
  return __builtin_amdgcn_wmma_f32_16x16x32_bf16(false, a, false, b, (short)0, c, false, false);
}

// ---------------------------------------------------------------- convert ---
__global__ void cvt_f32_bf16_kernel(const float* __restrict__ in,
                                    bf16_t* __restrict__ out, int n) {
  int i = blockIdx.x * blockDim.x + threadIdx.x;
  if (i < n) out[i] = (bf16_t)in[i];
}

// ------------------------------------------------- QKV GEMM + fused RoPE ---
// Each wave computes a 32x64 tile of one of Q/K/V = x @ W^T (bf16 in, f32 acc).
// Q,K stored [B,H,S,64] with RoPE applied; V stored transposed [B,H,64,S].
__global__ void __launch_bounds__(256)
qkv_rope_kernel(const bf16_t* __restrict__ xbf,
                const bf16_t* __restrict__ wq,
                const bf16_t* __restrict__ wk,
                const bf16_t* __restrict__ wv,
                const int*    __restrict__ tokpos,
                bf16_t* __restrict__ Q,
                bf16_t* __restrict__ K,
                bf16_t* __restrict__ Vt) {
  const int MT = BS_ROWS / 32;           // 256
  const int NT = DMODEL / 64;            // 16
  int wid = blockIdx.x * (blockDim.x >> 5) + (threadIdx.x >> 5);
  int mat = wid / (MT * NT);             // 0=Q 1=K 2=V
  int rem = wid % (MT * NT);
  int mt = rem / NT, nt = rem % NT;
  int m0 = mt * 32, n0 = nt * 64;
  const bf16_t* W = (mat == 0) ? wq : (mat == 1 ? wk : wv);

  f32x8 acc[2][4];
#pragma unroll
  for (int im = 0; im < 2; ++im)
#pragma unroll
    for (int in = 0; in < 4; ++in)
#pragma unroll
      for (int r = 0; r < 8; ++r) acc[im][in][r] = 0.0f;

  for (int k0 = 0; k0 < DMODEL; k0 += 32) {
    // prefetch next A k-slab into caches
    __builtin_prefetch(xbf + (size_t)m0 * DMODEL + k0 + 32, 0, 1);
    bf16x16 a0 = load_a_frag(xbf, DMODEL, m0,      k0);
    bf16x16 a1 = load_a_frag(xbf, DMODEL, m0 + 16, k0);
#pragma unroll
    for (int in = 0; in < 4; ++in) {
      bf16x16 b = load_b_frag(W, DMODEL, n0 + in * 16, k0);
      acc[0][in] = wmma_bf16(a0, b, acc[0][in]);
      acc[1][in] = wmma_bf16(a1, b, acc[1][in]);
    }
  }

  const int lane = threadIdx.x & 31;
  const int nlo  = lane & 15;
  const int hlf  = lane >> 4;
  const float LOG1E4_OVER_HD = 9.210340371976184f / (float)HDIM; // ln(10000)/64

#pragma unroll
  for (int im = 0; im < 2; ++im) {
#pragma unroll
    for (int r = 0; r < 8; ++r) {
      int g = m0 + im * 16 + r + 8 * hlf;      // global row in [0, 8192)
      int b = g >> 11;                          // / 2048
      int s = g & (S_LEN - 1);
#pragma unroll
      for (int in = 0; in < 4; ++in) {
        int ng = n0 + in * 16 + nlo;            // column in [0,1024)
        int h  = ng >> 6;
        int dd = ng & (HDIM - 1);
        float v = acc[im][in][r];
        if (mat == 2) {
          // V^T: [B,H,64,S]
          Vt[((size_t)(b * NHEADS + h) * HDIM + dd) * S_LEN + s] = (bf16_t)v;
        } else {
          // RoPE: partner element dd^1 lives in lane^1 (same row, same half)
          float part = __shfl_xor(v, 1, 32);
          float pos  = (float)tokpos[s];
          float invf = __expf(-(float)(dd & ~1) * LOG1E4_OVER_HD);
          float ang  = pos * invf;
          float sn, cs;
          __sincosf(ang, &sn, &cs);
          float res = ((dd & 1) == 0) ? (cs * v - sn * part)
                                      : (sn * part + cs * v);
          bf16_t* dst = (mat == 0) ? Q : K;
          dst[((size_t)(b * NHEADS + h) * S_LEN + s) * HDIM + dd] = (bf16_t)res;
        }
      }
    }
  }
}

// ----------------------------------------------------- flash attention ------
// One wave owns a 32-row q-tile of one (b,h). Online softmax, causal.
__global__ void __launch_bounds__(128)
attn_kernel(const bf16_t* __restrict__ Q,
            const bf16_t* __restrict__ K,
            const bf16_t* __restrict__ Vt,
            bf16_t* __restrict__ O) {
  __shared__ __align__(16) bf16_t plds[4][32 * 40];   // P staging per wave

  const int wv  = threadIdx.x >> 5;
  int wid = blockIdx.x * 4 + wv;
  int bh  = wid >> 6;                       // / (2048/32)
  int qt  = wid & 63;
  int b   = bh >> 4, h = bh & 15;
  int q0  = qt * 32;
  const bf16_t* Qh = Q  + (size_t)bh * S_LEN * HDIM;
  const bf16_t* Kh = K  + (size_t)bh * S_LEN * HDIM;
  const bf16_t* Vh = Vt + (size_t)bh * HDIM * S_LEN;
  const int lane = threadIdx.x & 31;
  const int nlo  = lane & 15;
  const int hlf  = lane >> 4;
  bf16_t* my = plds[wv];

  // Q tile as A fragments: 32 rows x 64 (two 16-row, two 32-K fragments)
  bf16x16 aq[2][2];
#pragma unroll
  for (int im = 0; im < 2; ++im)
#pragma unroll
    for (int kt = 0; kt < 2; ++kt)
      aq[im][kt] = load_a_frag(Qh, HDIM, q0 + im * 16, kt * 32);

  f32x8 o_acc[2][4];
  float m_run[2][8], l_run[2][8];
#pragma unroll
  for (int im = 0; im < 2; ++im) {
#pragma unroll
    for (int in = 0; in < 4; ++in)
#pragma unroll
      for (int r = 0; r < 8; ++r) o_acc[im][in][r] = 0.0f;
#pragma unroll
    for (int r = 0; r < 8; ++r) { m_run[im][r] = -1e30f; l_run[im][r] = 0.0f; }
  }

  for (int kb = 0; kb <= qt; ++kb) {
    const int kk0 = kb * 32;
    // scores S = Q K^T  (M=q 32, N=kpos 32, K=dd 64)
    f32x8 sc[2][2];
#pragma unroll
    for (int im = 0; im < 2; ++im)
#pragma unroll
      for (int in = 0; in < 2; ++in)
#pragma unroll
        for (int r = 0; r < 8; ++r) sc[im][in][r] = 0.0f;
#pragma unroll
    for (int kt = 0; kt < 2; ++kt) {
      bf16x16 bk0 = load_b_frag(Kh, HDIM, kk0,      kt * 32);
      bf16x16 bk1 = load_b_frag(Kh, HDIM, kk0 + 16, kt * 32);
      sc[0][0] = wmma_bf16(aq[0][kt], bk0, sc[0][0]);
      sc[0][1] = wmma_bf16(aq[0][kt], bk1, sc[0][1]);
      sc[1][0] = wmma_bf16(aq[1][kt], bk0, sc[1][0]);
      sc[1][1] = wmma_bf16(aq[1][kt], bk1, sc[1][1]);
    }
    // scale + causal mask (diagonal block only)
    const bool diag = (kb == qt);
#pragma unroll
    for (int im = 0; im < 2; ++im)
#pragma unroll
      for (int in = 0; in < 2; ++in)
#pragma unroll
        for (int r = 0; r < 8; ++r) {
          float v = sc[im][in][r] * 0.125f;   // 1/sqrt(64)
          if (diag) {
            int qrow = im * 16 + r + 8 * hlf;
            int kcol = in * 16 + nlo;
            if (kcol > qrow) v = -1e30f;
          }
          sc[im][in][r] = v;
        }
    // online softmax update (row m lives in one 16-lane half)
#pragma unroll
    for (int im = 0; im < 2; ++im)
#pragma unroll
      for (int r = 0; r < 8; ++r) {
        float bm = fmaxf(sc[im][0][r], sc[im][1][r]);
#pragma unroll
        for (int d = 1; d < 16; d <<= 1) bm = fmaxf(bm, __shfl_xor(bm, d, 32));
        float mn    = fmaxf(m_run[im][r], bm);
        float alpha = __expf(m_run[im][r] - mn);
        m_run[im][r] = mn;
        float rs = 0.0f;
#pragma unroll
        for (int in = 0; in < 2; ++in) {
          float p = __expf(sc[im][in][r] - mn);
          sc[im][in][r] = p;
          rs += p;
        }
#pragma unroll
        for (int d = 1; d < 16; d <<= 1) rs += __shfl_xor(rs, d, 32);
        l_run[im][r] = l_run[im][r] * alpha + rs;
#pragma unroll
        for (int in = 0; in < 4; ++in) o_acc[im][in][r] *= alpha;
      }
    // P (D-layout) -> LDS row-major [32][40], re-read as A fragments
#pragma unroll
    for (int im = 0; im < 2; ++im)
#pragma unroll
      for (int in = 0; in < 2; ++in)
#pragma unroll
        for (int r = 0; r < 8; ++r)
          my[(im * 16 + r + 8 * hlf) * 40 + in * 16 + nlo] = (bf16_t)sc[im][in][r];
    asm volatile("s_wait_dscnt 0" ::: "memory");
    bf16x16 ap[2];
#pragma unroll
    for (int im = 0; im < 2; ++im)
      ap[im] = load_a_frag(my, 40, im * 16, 0);
    // O += P @ V  (B from V^T, rows = dd, contiguous kpos)
#pragma unroll
    for (int in = 0; in < 4; ++in) {
      bf16x16 bv = load_b_frag(Vh, S_LEN, in * 16, kk0);
      o_acc[0][in] = wmma_bf16(ap[0], bv, o_acc[0][in]);
      o_acc[1][in] = wmma_bf16(ap[1], bv, o_acc[1][in]);
    }
  }

  // normalize + store to O [B,S,1024] (heads re-interleaved), bf16
#pragma unroll
  for (int im = 0; im < 2; ++im)
#pragma unroll
    for (int r = 0; r < 8; ++r) {
      float inv_l = 1.0f / l_run[im][r];
      int q = q0 + im * 16 + r + 8 * hlf;
#pragma unroll
      for (int in = 0; in < 4; ++in) {
        int dd = in * 16 + nlo;
        O[((size_t)(b * S_LEN + q)) * DMODEL + h * HDIM + dd] =
            (bf16_t)(o_acc[im][in][r] * inv_l);
      }
    }
}

// ---------------------------------------------------- output projection -----
__global__ void __launch_bounds__(256)
out_proj_kernel(const bf16_t* __restrict__ Obf,
                const bf16_t* __restrict__ wo,
                float* __restrict__ out) {
  const int NT = DMODEL / 64;            // 16
  int wid = blockIdx.x * (blockDim.x >> 5) + (threadIdx.x >> 5);
  int mt = wid / NT, nt = wid % NT;
  int m0 = mt * 32, n0 = nt * 64;

  f32x8 acc[2][4];
#pragma unroll
  for (int im = 0; im < 2; ++im)
#pragma unroll
    for (int in = 0; in < 4; ++in)
#pragma unroll
      for (int r = 0; r < 8; ++r) acc[im][in][r] = 0.0f;

  for (int k0 = 0; k0 < DMODEL; k0 += 32) {
    __builtin_prefetch(Obf + (size_t)m0 * DMODEL + k0 + 32, 0, 1);
    bf16x16 a0 = load_a_frag(Obf, DMODEL, m0,      k0);
    bf16x16 a1 = load_a_frag(Obf, DMODEL, m0 + 16, k0);
#pragma unroll
    for (int in = 0; in < 4; ++in) {
      bf16x16 b = load_b_frag(wo, DMODEL, n0 + in * 16, k0);
      acc[0][in] = wmma_bf16(a0, b, acc[0][in]);
      acc[1][in] = wmma_bf16(a1, b, acc[1][in]);
    }
  }

  const int lane = threadIdx.x & 31;
  const int nlo  = lane & 15;
  const int hlf  = lane >> 4;
#pragma unroll
  for (int im = 0; im < 2; ++im)
#pragma unroll
    for (int r = 0; r < 8; ++r) {
      int g = m0 + im * 16 + r + 8 * hlf;
#pragma unroll
      for (int in = 0; in < 4; ++in) {
        int ng = n0 + in * 16 + nlo;
        out[(size_t)g * DMODEL + ng] = acc[im][in][r];
      }
    }
}

// ---------------------------------------------------------------- launch ----
extern "C" void kernel_launch(void* const* d_in, const int* in_sizes, int n_in,
                              void* d_out, int out_size, void* d_ws, size_t ws_size,
                              hipStream_t stream) {
  const float* x   = (const float*)d_in[0];
  const float* Wq  = (const float*)d_in[1];
  const float* Wk  = (const float*)d_in[2];
  const float* Wv  = (const float*)d_in[3];
  const float* Wo  = (const float*)d_in[4];
  const int*   pos = (const int*)d_in[5];
  float* out = (float*)d_out;

  bf16_t* ws = (bf16_t*)d_ws;
  const size_t XN = (size_t)BS_ROWS * DMODEL;     // 8M elems
  const size_t WN = (size_t)DMODEL * DMODEL;      // 1M elems
  size_t off = 0;
  bf16_t* xbf = ws + off; off += XN;
  bf16_t* wqb = ws + off; off += WN;
  bf16_t* wkb = ws + off; off += WN;
  bf16_t* wvb = ws + off; off += WN;
  bf16_t* wob = ws + off; off += WN;
  bf16_t* Qb  = ws + off; off += XN;   // [B,H,S,64]
  bf16_t* Kb  = ws + off; off += XN;   // [B,H,S,64]
  bf16_t* Vtb = ws + off; off += XN;   // [B,H,64,S]
  bf16_t* Ob  = ws + off; off += XN;   // [B,S,1024]

  cvt_f32_bf16_kernel<<<(int)((XN + 255) / 256), 256, 0, stream>>>(x,  xbf, (int)XN);
  cvt_f32_bf16_kernel<<<(int)((WN + 255) / 256), 256, 0, stream>>>(Wq, wqb, (int)WN);
  cvt_f32_bf16_kernel<<<(int)((WN + 255) / 256), 256, 0, stream>>>(Wk, wkb, (int)WN);
  cvt_f32_bf16_kernel<<<(int)((WN + 255) / 256), 256, 0, stream>>>(Wv, wvb, (int)WN);
  cvt_f32_bf16_kernel<<<(int)((WN + 255) / 256), 256, 0, stream>>>(Wo, wob, (int)WN);

  // 3 mats * 256 m-tiles * 16 n-tiles = 12288 waves / 8 per block
  qkv_rope_kernel<<<1536, 256, 0, stream>>>(xbf, wqb, wkb, wvb, pos, Qb, Kb, Vtb);
  // 4*16 heads * 64 q-tiles = 4096 waves / 4 per block
  attn_kernel<<<1024, 128, 0, stream>>>(Qb, Kb, Vtb, Ob);
  // 256*16 = 4096 waves / 8 per block
  out_proj_kernel<<<512, 256, 0, stream>>>(Ob, wob, out);
}
